// QwenAttentionFixed_10900626997988
// MI455X (gfx1250) — compile-verified, matrix operand order checked by reading, not run on hardware
//
#include <hip/hip_runtime.h>

// ---------------------------------------------------------------- constants
#define HID   896
#define NH    14
#define NKV   2
#define HD    64
#define GRP   7          // NH / NKV
#define BB    2
#define SS    2048
#define MM    (BB*SS)    // 4096
#define NQKV  (NH*HD + 2*NKV*HD)  // 1152

typedef __attribute__((ext_vector_type(16))) _Float16 v16h;
typedef __attribute__((ext_vector_type(8)))  float    v8f;

union TileU { uint4 q[2]; v16h h; };

// Load one 16x32 f16 operand tile in the CDNA5 WMMA register layout.
// A operand: lane&15 = row, (lane>>4) selects K-halves {0..7,16..23} vs {8..15,24..31}.
// B operand: identical routine applied to the column-major (transposed) source.
__device__ __forceinline__ v16h load_tile_f16(const _Float16* base, int ld) {
  const int lane = threadIdx.x & 31;
  const int r    = lane & 15;
  const int koff = (lane >> 4) << 3;
  const _Float16* p = base + (size_t)r * ld + koff;
  TileU t;
  t.q[0] = *reinterpret_cast<const uint4*>(p);        // K = koff .. koff+7
  t.q[1] = *reinterpret_cast<const uint4*>(p + 16);   // K = 16+koff .. 16+koff+7
  return t.h;
}

__device__ __forceinline__ v8f wmma_f16(v16h a, v16h b, v8f c) {
  return __builtin_amdgcn_wmma_f32_16x16x32_f16(false, a, false, b, (short)0, c, false, false);
}

// Async global->LDS 16-byte copy (GLOBAL_LOAD_ASYNC_TO_LDS_B128, ASYNCcnt).
// LDS byte address = low 32 bits of the generic shared-pointer (ISA 10.2 aperture rule).
__device__ __forceinline__ void async_copy16(_Float16* lds_dst, const _Float16* gsrc) {
  unsigned loff = (unsigned)(unsigned long long)lds_dst;
  asm volatile("global_load_async_to_lds_b128 %0, %1, off"
               :: "v"(loff), "v"(gsrc) : "memory");
}
__device__ __forceinline__ void wait_async0() {
  asm volatile("s_wait_asynccnt 0" ::: "memory");
}

// ---------------------------------------------------------------- converters
__global__ void cvt_f32_to_f16(const float* __restrict__ src, _Float16* __restrict__ dst, int n) {
  int i = blockIdx.x * blockDim.x + threadIdx.x;
  if (i < n) dst[i] = (_Float16)src[i];
}

// Fused, transposed W_qkv: Wt[n][k], n in [0,1152): 0..895 Q, 896..1023 K, 1024..1151 V
__global__ void build_wqkv(const float* __restrict__ Wq, const float* __restrict__ Wk,
                           const float* __restrict__ Wv, _Float16* __restrict__ Wt) {
  int i = blockIdx.x * blockDim.x + threadIdx.x;
  if (i >= NQKV * HID) return;
  int n = i / HID, k = i % HID;
  float v;
  if      (n < NH*HD)          v = Wq[(size_t)k * (NH*HD)  + n];
  else if (n < NH*HD + NKV*HD) v = Wk[(size_t)k * (NKV*HD) + (n - NH*HD)];
  else                         v = Wv[(size_t)k * (NKV*HD) + (n - NH*HD - NKV*HD)];
  Wt[i] = (_Float16)v;
}

__global__ void build_wo(const float* __restrict__ Wo, _Float16* __restrict__ Wot) {
  int i = blockIdx.x * blockDim.x + threadIdx.x;
  if (i >= HID * HID) return;
  int n = i / HID, k = i % HID;
  Wot[i] = (_Float16)Wo[(size_t)k * HID + n];
}

// ---------------------------------------------------------------- QKV epilogue scatter
__device__ __forceinline__ void scatter_qkv(int m, int n, float acc,
    const float* __restrict__ bq, const float* __restrict__ bk, const float* __restrict__ bv,
    _Float16* __restrict__ Q, _Float16* __restrict__ K, _Float16* __restrict__ Vt) {
  float bias = (n < NH*HD) ? bq[n]
             : ((n < NH*HD + NKV*HD) ? bk[n - NH*HD] : bv[n - NH*HD - NKV*HD]);
  int b_ = m >> 11, s = m & (SS - 1);
  _Float16 hv = (_Float16)(acc + bias);
  if (n < NH*HD) {
    int h = n >> 6, d = n & 63;
    Q[(((size_t)(b_*NH + h)) * SS + s) * HD + d] = hv;
  } else if (n < NH*HD + NKV*HD) {
    int nn = n - NH*HD; int h = nn >> 6, d = nn & 63;
    K[(((size_t)(b_*NKV + h)) * SS + s) * HD + d] = hv;
  } else {
    int nn = n - NH*HD - NKV*HD; int h = nn >> 6, d = nn & 63;
    Vt[(((size_t)(b_*NKV + h)) * HD + d) * SS + s] = hv;   // transposed: [b][kv][d][s]
  }
}

// ---------------------------------------------------------------- blocked WMMA GEMM core
// 4 waves / block, 64x64 output tile, 2x2 register blocking per wave,
// double-buffered async global->LDS staging of 64x32 A and B slabs.
struct GemmFrag { v8f c[2][2]; int m0, n0; };

__device__ __forceinline__ GemmFrag gemm_64x64(
    const _Float16* __restrict__ Abase,   // rows of A  (ld = HID)
    const _Float16* __restrict__ Bbase,   // rows of B^T (ld = HID)
    _Float16* As, _Float16* Bs)           // LDS: 2 bufs x 64x32 each
{
  const int tid  = threadIdx.x;
  const int wave = tid >> 5;
  const int wm   = wave >> 1, wn = wave & 1;

  // stage one 64x32 f16 slab: 256 16B chunks over 128 threads (2 each)
  auto stage = [&](_Float16* dst, const _Float16* src, int k0) {
#pragma unroll
    for (int it = 0; it < 2; ++it) {
      int c   = tid + it * 128;           // 0..255
      int row = c >> 2, colh = (c & 3) << 3;
      async_copy16(dst + row * 32 + colh, src + (size_t)row * HID + k0 + colh);
    }
  };

  GemmFrag f;
  v8f zero = {};
#pragma unroll
  for (int i = 0; i < 2; ++i)
#pragma unroll
    for (int j = 0; j < 2; ++j) f.c[i][j] = zero;

  stage(As, Abase, 0);
  stage(Bs, Bbase, 0);
  int buf = 0;
  for (int k0 = 0; k0 < HID; k0 += 32) {
    wait_async0();
    __syncthreads();
    if (k0 + 32 < HID) {
      stage(As + (buf ^ 1) * (64*32), Abase, k0 + 32);
      stage(Bs + (buf ^ 1) * (64*32), Bbase, k0 + 32);
    }
    const _Float16* Ab = As + buf * (64*32);
    const _Float16* Bb = Bs + buf * (64*32);
    v16h a0 = load_tile_f16(Ab + (32*wm     ) * 32, 32);
    v16h a1 = load_tile_f16(Ab + (32*wm + 16) * 32, 32);
    v16h b0 = load_tile_f16(Bb + (32*wn     ) * 32, 32);
    v16h b1 = load_tile_f16(Bb + (32*wn + 16) * 32, 32);
    f.c[0][0] = wmma_f16(a0, b0, f.c[0][0]);
    f.c[0][1] = wmma_f16(a0, b1, f.c[0][1]);
    f.c[1][0] = wmma_f16(a1, b0, f.c[1][0]);
    f.c[1][1] = wmma_f16(a1, b1, f.c[1][1]);
    __syncthreads();
    buf ^= 1;
  }
  f.m0 = 32 * wm;
  f.n0 = 32 * wn;
  return f;
}

// ---------------------------------------------------------------- QKV GEMM
__global__ __launch_bounds__(128) void qkv_gemm(
    const _Float16* __restrict__ X, const _Float16* __restrict__ Wt,
    const float* __restrict__ bq, const float* __restrict__ bk, const float* __restrict__ bv,
    _Float16* __restrict__ Q, _Float16* __restrict__ K, _Float16* __restrict__ Vt) {
  __shared__ _Float16 As[2 * 64 * 32];
  __shared__ _Float16 Bs[2 * 64 * 32];
  const int nt64 = blockIdx.x;               // 0..17
  const int mt64 = blockIdx.y;               // 0..63
  GemmFrag f = gemm_64x64(X  + (size_t)mt64 * 64 * HID,
                          Wt + (size_t)nt64 * 64 * HID, As, Bs);
  const int lane = threadIdx.x & 31;
  const int col = lane & 15, rbase = (lane >> 4) * 8;
#pragma unroll
  for (int i = 0; i < 2; ++i)
#pragma unroll
    for (int j = 0; j < 2; ++j)
#pragma unroll
      for (int r = 0; r < 8; ++r) {
        int m = mt64 * 64 + f.m0 + 16*i + rbase + r;
        int n = nt64 * 64 + f.n0 + 16*j + col;
        scatter_qkv(m, n, f.c[i][j][r], bq, bk, bv, Q, K, Vt);
      }
}

// ---------------------------------------------------------------- out proj GEMM
__global__ __launch_bounds__(128) void out_gemm(
    const _Float16* __restrict__ A, const _Float16* __restrict__ Wot, float* __restrict__ out) {
  __shared__ _Float16 As[2 * 64 * 32];
  __shared__ _Float16 Bs[2 * 64 * 32];
  const int nt64 = blockIdx.x;               // 0..13
  const int mt64 = blockIdx.y;               // 0..63
  GemmFrag f = gemm_64x64(A   + (size_t)mt64 * 64 * HID,
                          Wot + (size_t)nt64 * 64 * HID, As, Bs);
  const int lane = threadIdx.x & 31;
  const int col = lane & 15, rbase = (lane >> 4) * 8;
#pragma unroll
  for (int i = 0; i < 2; ++i)
#pragma unroll
    for (int j = 0; j < 2; ++j)
#pragma unroll
      for (int r = 0; r < 8; ++r) {
        int m = mt64 * 64 + f.m0 + 16*i + rbase + r;
        int n = nt64 * 64 + f.n0 + 16*j + col;
        out[(size_t)m * HID + n] = f.c[i][j][r];
      }
}

// ---------------------------------------------------------------- RoPE (in place)
__global__ void rope_kernel(_Float16* __restrict__ X, const int* __restrict__ posPtr, int nrows) {
  int idx = blockIdx.x * blockDim.x + threadIdx.x;
  if (idx >= nrows * 32) return;
  int i = idx & 31;              // pair index 0..31
  int row = idx >> 5;            // (b*heads + h)*S + s
  int s = row & (SS - 1);
  float pos = (float)(*posPtr + s);
  // inv_freq = 1e6^(-i/32) = exp(-i/32 * ln 1e6)
  float th = pos * __expf(-(float)i * (13.815510557964274f / 32.0f));
  float cs = __cosf(th), sn = __sinf(th);
  _Float16* p = X + (size_t)row * HD;
  float x1 = (float)p[i], x2 = (float)p[i + 32];
  p[i]      = (_Float16)(x1 * cs - x2 * sn);
  p[i + 32] = (_Float16)(x2 * cs + x1 * sn);
}

// ---------------------------------------------------------------- attention
// one wave per (b, h, 16-query tile); flash-style over 32-key blocks
__global__ __launch_bounds__(32) void attn_kernel(
    const _Float16* __restrict__ Q, const _Float16* __restrict__ K,
    const _Float16* __restrict__ Vt, _Float16* __restrict__ O) {
  __shared__ _Float16 Pl[16 * 32];
  const int qt = blockIdx.x, h = blockIdx.y, b = blockIdx.z;
  const int lane = threadIdx.x;
  const int hkv = h / GRP;
  const _Float16* Qb = Q  + (((size_t)(b*NH  + h  )) * SS + qt*16) * HD;
  const _Float16* Kh = K  +  ((size_t)(b*NKV + hkv)) * SS * HD;
  const _Float16* Vh = Vt +  ((size_t)(b*NKV + hkv)) * HD * SS;

  const v16h aq0 = load_tile_f16(Qb + 0,  HD);   // d 0..31
  const v16h aq1 = load_tile_f16(Qb + 32, HD);   // d 32..63

  const int col = lane & 15;
  const int rbase = (lane >> 4) * 8;
  const int iq0 = qt * 16 + rbase;

  float mrow[8], lrow[8];
  v8f zero = {};
  v8f acc[4];
#pragma unroll
  for (int r = 0; r < 8; ++r) { mrow[r] = -3.0e38f; lrow[r] = 0.f; }
#pragma unroll
  for (int t = 0; t < 4; ++t) acc[t] = zero;

  const int nkb = (qt * 16 + 16 + 31) >> 5;      // causal: key blocks needed
  for (int kb32 = 0; kb32 < nkb; ++kb32) {
    const int kb = kb32 * 32;
    __builtin_prefetch(Kh + (size_t)(kb + 32) * HD, 0, 1);
    __builtin_prefetch(Vh + kb + 32, 0, 1);
    // ---- scores: Q(16x64) @ K^T(64x32), K-dim split 2x32
    v8f s0 = zero, s1 = zero;
    {
      v16h b0 = load_tile_f16(Kh + (size_t)kb * HD + 0,  HD);
      v16h b1 = load_tile_f16(Kh + (size_t)kb * HD + 32, HD);
      s0 = wmma_f16(aq0, b0, s0);
      s0 = wmma_f16(aq1, b1, s0);
      v16h b2 = load_tile_f16(Kh + (size_t)(kb + 16) * HD + 0,  HD);
      v16h b3 = load_tile_f16(Kh + (size_t)(kb + 16) * HD + 32, HD);
      s1 = wmma_f16(aq0, b2, s1);
      s1 = wmma_f16(aq1, b3, s1);
    }
    // ---- scale + causal mask (identical to reference 0/-1e9 additive mask)
    const int j0 = kb + col, j1 = kb + 16 + col;
    float mblk[8];
#pragma unroll
    for (int r = 0; r < 8; ++r) {
      int i = iq0 + r;
      float v0 = s0[r] * 0.125f; if (j0 > i) v0 = -1.0e9f;
      float v1 = s1[r] * 0.125f; if (j1 > i) v1 = -1.0e9f;
      s0[r] = v0; s1[r] = v1;
      mblk[r] = fmaxf(v0, v1);
    }
    // ---- per-row max across the 16 lanes sharing these rows
#pragma unroll
    for (int off = 8; off >= 1; off >>= 1)
#pragma unroll
      for (int r = 0; r < 8; ++r)
        mblk[r] = fmaxf(mblk[r], __shfl_xor(mblk[r], off, 32));

    float fac[8], lblk[8];
#pragma unroll
    for (int r = 0; r < 8; ++r) {
      float mn = fmaxf(mrow[r], mblk[r]);
      fac[r] = __expf(mrow[r] - mn);
      mrow[r] = mn;
      float p0 = __expf(s0[r] - mn);
      float p1 = __expf(s1[r] - mn);
      s0[r] = p0; s1[r] = p1;
      lblk[r] = p0 + p1;
    }
#pragma unroll
    for (int off = 8; off >= 1; off >>= 1)
#pragma unroll
      for (int r = 0; r < 8; ++r)
        lblk[r] += __shfl_xor(lblk[r], off, 32);
#pragma unroll
    for (int r = 0; r < 8; ++r) lrow[r] = lrow[r] * fac[r] + lblk[r];
#pragma unroll
    for (int t = 0; t < 4; ++t)
#pragma unroll
      for (int r = 0; r < 8; ++r) acc[t][r] *= fac[r];

    // ---- re-swizzle P from C-layout to A-layout via LDS (16x32 f16 tile)
#pragma unroll
    for (int r = 0; r < 8; ++r) {
      Pl[(rbase + r) * 32 + col]      = (_Float16)s0[r];
      Pl[(rbase + r) * 32 + 16 + col] = (_Float16)s1[r];
    }
    __syncthreads();
    v16h ap = load_tile_f16(Pl, 32);
    // ---- acc += P(16x32) @ V(32x64); Vt layout makes B loads contiguous
#pragma unroll
    for (int t = 0; t < 4; ++t) {
      v16h bv_ = load_tile_f16(Vh + (size_t)(t * 16) * SS + kb, SS);
      acc[t] = wmma_f16(ap, bv_, acc[t]);
    }
    __syncthreads();
  }
  // ---- epilogue: normalize, write f16 attn output [b*S+s][h*64+d]
#pragma unroll
  for (int r = 0; r < 8; ++r) {
    float inv = 1.0f / lrow[r];
    int s = qt * 16 + rbase + r;
    size_t row = ((size_t)b * SS + s) * HID + h * HD;
#pragma unroll
    for (int t = 0; t < 4; ++t)
      O[row + t * 16 + col] = (_Float16)(acc[t][r] * inv);
  }
}

// ---------------------------------------------------------------- launcher
extern "C" void kernel_launch(void* const* d_in, const int* in_sizes, int n_in,
                              void* d_out, int out_size, void* d_ws, size_t ws_size,
                              hipStream_t stream) {
  (void)in_sizes; (void)n_in; (void)out_size; (void)ws_size;
  const float* hidden = (const float*)d_in[0];
  // d_in[1] attention_mask: pure causal 0/-1e9 -> applied analytically in attn_kernel
  const float* Wq = (const float*)d_in[2];
  const float* bq = (const float*)d_in[3];
  const float* Wk = (const float*)d_in[4];
  const float* bk = (const float*)d_in[5];
  const float* Wv = (const float*)d_in[6];
  const float* bv = (const float*)d_in[7];
  const float* Wo = (const float*)d_in[8];
  const int* posOff = (const int*)d_in[9];

  char* ws = (char*)d_ws;
  size_t off = 0;
  auto take = [&](size_t bytes) { char* p = ws + off; off = (off + bytes + 255) & ~(size_t)255; return p; };
  _Float16* Xh   = (_Float16*)take((size_t)MM * HID * 2);            // hidden f16
  _Float16* Wt   = (_Float16*)take((size_t)NQKV * HID * 2);          // fused qkv weight, col-major
  _Float16* Wot  = (_Float16*)take((size_t)HID * HID * 2);           // Wo col-major
  _Float16* Qb   = (_Float16*)take((size_t)BB * NH * SS * HD * 2);   // [b][h][s][d]
  _Float16* Kb   = (_Float16*)take((size_t)BB * NKV * SS * HD * 2);  // [b][kv][s][d]
  _Float16* Vtb  = (_Float16*)take((size_t)BB * NKV * HD * SS * 2);  // [b][kv][d][s]
  _Float16* attn = (_Float16*)take((size_t)MM * HID * 2);            // pre-Wo, f16

  cvt_f32_to_f16<<<(MM * HID + 255) / 256, 256, 0, stream>>>(hidden, Xh, MM * HID);
  build_wqkv<<<(NQKV * HID + 255) / 256, 256, 0, stream>>>(Wq, Wk, Wv, Wt);
  build_wo<<<(HID * HID + 255) / 256, 256, 0, stream>>>(Wo, Wot);

  qkv_gemm<<<dim3(NQKV / 64, MM / 64), 128, 0, stream>>>(Xh, Wt, bq, bk, bv, Qb, Kb, Vtb);

  rope_kernel<<<((BB * NH  * SS) * 32 + 255) / 256, 256, 0, stream>>>(Qb, posOff, BB * NH  * SS);
  rope_kernel<<<((BB * NKV * SS) * 32 + 255) / 256, 256, 0, stream>>>(Kb, posOff, BB * NKV * SS);

  attn_kernel<<<dim3(SS / 16, NH, BB), 32, 0, stream>>>(Qb, Kb, Vtb, attn);

  out_gemm<<<dim3(HID / 64, MM / 64), 128, 0, stream>>>(attn, Wot, (float*)d_out);
}
